// TGN_72052371357797
// MI455X (gfx1250) — matrix operand order repaired and checked by reading, not compiled, not run on hardware
//
#include <hip/hip_runtime.h>
#include <hip/hip_bf16.h>

// ---------------------------------------------------------------------------
// Single-query MHA (TGN attention_node) for MI455X / gfx1250, wave32 + WMMA.
//
// Algebraic restructure (1 query/row): bk drops out of softmax; bv folds to a
// constant since softmax weights sum to 1.  Pipeline:
//   prep:   bf16 weights (WkT transposed), bf16 source
//   G1:     q_s  = (src @ Wq^T + bq) * hd^-0.5                  [WMMA bf16]
//   G2(h):  r_h  = q_s[:,h] @ Wk_h                              [WMMA bf16]
//   A:      scores = gene·r ; masked softmax ; u = w^T·gene ; attn_weight
//   G3(h):  attn_h = u_h @ Wv_h^T + bv_h                        [WMMA bf16]
//   G4:     out = gate((attn @ Wout^T + bout)) * mask_n + src   [WMMA bf16]
// ---------------------------------------------------------------------------

typedef __bf16 v16bf __attribute__((ext_vector_type(16)));
typedef float  v8f   __attribute__((ext_vector_type(8)));

union FragBF { uint4 u[2]; v16bf v; };

__device__ __forceinline__ unsigned short f2bf(float f) {
    unsigned u = __float_as_uint(f);
    u += 0x7fffu + ((u >> 16) & 1u);          // round-to-nearest-even
    return (unsigned short)(u >> 16);
}
__device__ __forceinline__ float bf2f(unsigned short b) {
    return __uint_as_float(((unsigned)b) << 16);
}

// ------------------------------ prep kernels -------------------------------
__global__ void cvt_bf16_kernel(const float* __restrict__ src,
                                unsigned short* __restrict__ dst, int n) {
    int i = blockIdx.x * 256 + threadIdx.x;
    if (i < n) dst[i] = f2bf(src[i]);
}

// WkT[g][c] = Wk[c][g], both 512x512.  32x32 LDS tiles, 256 threads (32x8).
__global__ __launch_bounds__(256) void transpose_bf16_kernel(
    const float* __restrict__ src, unsigned short* __restrict__ dst) {
    __shared__ float t[32][33];
    const int bx = blockIdx.x & 15, by = blockIdx.x >> 4;
    const int lx = threadIdx.x & 31, ly = threadIdx.x >> 5;
    for (int i = 0; i < 32; i += 8)
        t[ly + i][lx] = src[(size_t)(by * 32 + ly + i) * 512 + bx * 32 + lx];
    __syncthreads();
    for (int i = 0; i < 32; i += 8)
        dst[(size_t)(bx * 32 + ly + i) * 512 + by * 32 + lx] = f2bf(t[lx][ly + i]);
}

// ------------------------------ WMMA GEMM ----------------------------------
// C[row, col] = sum_k A[row, k] * Bw[col*ldb + k]   (Bw row-major per column)
// Block: 256 thr (8 waves). Tile: 32 rows x 256 cols. Grid: (B/32, N/256).
// EPI 0: bf16 out, (acc + bias)*scale.  EPI 1: fp32 out with gate+residual.
template <int K, int EPI>
__global__ __launch_bounds__(256) void gemm_bf16_kernel(
    const unsigned short* __restrict__ A, int lda,
    const unsigned short* __restrict__ Bw, int ldb,
    const float* __restrict__ bias, float scale,
    unsigned short* __restrict__ Cbf, int ldc,
    float* __restrict__ Cf,
    const float* __restrict__ src, const float* __restrict__ mask_n,
    const int* __restrict__ invalid) {
    constexpr int KP = K + 8;                     // LDS pad: +16B per row
    __shared__ unsigned short Alds[32 * KP];

    const int tid  = threadIdx.x;
    const int wave = tid >> 5, lane = tid & 31;
    const int hi = lane >> 4, lx = lane & 15;
    const int row0 = blockIdx.x * 32;

    // stage 32 x K bf16 A tile -> LDS (16B chunks, fully coalesced)
    constexpr int CHROW = K / 8;
    for (int i = tid; i < 32 * CHROW; i += 256) {
        int r = i / CHROW, ch = i - r * CHROW;
        *(uint4*)&Alds[r * KP + ch * 8] =
            *(const uint4*)&A[(size_t)(row0 + r) * lda + ch * 8];
    }
    __syncthreads();

    v8f acc[2][2];
#pragma unroll
    for (int mt = 0; mt < 2; ++mt)
#pragma unroll
        for (int j = 0; j < 2; ++j) acc[mt][j] = v8f{0, 0, 0, 0, 0, 0, 0, 0};

    const int ncol0 = blockIdx.y * 256 + wave * 32;

    for (int kt = 0; kt < K / 32; ++kt) {
        FragBF a[2];
#pragma unroll
        for (int mt = 0; mt < 2; ++mt) {
            // A 16x32 frag: lanes 0-15 row m K{0-7,16-23}; lanes 16-31 K{8-15,24-31}
            const unsigned short* p = &Alds[(mt * 16 + lx) * KP + kt * 32 + hi * 8];
            a[mt].u[0] = *(const uint4*)p;
            a[mt].u[1] = *(const uint4*)(p + 16);
        }
#pragma unroll
        for (int j = 0; j < 2; ++j) {
            // B 32x16 frag: lane = column, lanes 0-15 K{0-7,16-23}; lanes 16-31 K{8-15,24-31}
            FragBF bfrag;
            const unsigned short* q =
                &Bw[(size_t)(ncol0 + j * 16 + lx) * ldb + kt * 32 + hi * 8];
            bfrag.u[0] = *(const uint4*)q;
            bfrag.u[1] = *(const uint4*)(q + 16);
#pragma unroll
            for (int mt = 0; mt < 2; ++mt)
                acc[mt][j] = __builtin_amdgcn_wmma_f32_16x16x32_bf16(
                    false, a[mt].v, false, bfrag.v, (short)0, acc[mt][j],
                    false, false);
        }
    }

    // epilogue: lane holds C[(hi?8:0)+r][lx] of each 16x16 tile
#pragma unroll
    for (int mt = 0; mt < 2; ++mt)
#pragma unroll
        for (int j = 0; j < 2; ++j) {
            const int col = ncol0 + j * 16 + lx;
            const float bs = bias ? bias[col] : 0.0f;
#pragma unroll
            for (int r = 0; r < 8; ++r) {
                const int row = row0 + mt * 16 + hi * 8 + r;
                const float v = acc[mt][j][r];
                if (EPI == 0) {
                    Cbf[(size_t)row * ldc + col] = f2bf((v + bs) * scale);
                } else {
                    const float t = invalid[row] ? 0.0f : (v + bs);
                    Cf[(size_t)row * 512 + col] =
                        t * mask_n[row] + src[(size_t)row * 512 + col];
                }
            }
        }
}

// --------------------- streaming attention (mem-bound) ---------------------
// One block per batch row: scores = gene·r, masked softmax, u = w^T gene.
__global__ __launch_bounds__(256) void attn_stream_kernel(
    const float* __restrict__ gene, const unsigned short* __restrict__ rws,
    const int* __restrict__ mask, unsigned short* __restrict__ uws,
    int* __restrict__ invalidws, float* __restrict__ attn_w) {
    extern __shared__ float smem[];
    float* geneS  = smem;                 // 32 x 516 (pad -> conflict-light)
    float* rS     = geneS + 32 * 516;     // 2 x 512
    float* wS     = rS + 1024;            // 2 x 32
    unsigned* mbx = (unsigned*)(wS + 64); // mask bits

    const int b = blockIdx.x, tid = threadIdx.x;
    const float* gb = gene + (size_t)b * 32 * 512;

    for (int i = tid; i < 4096; i += 256) {           // 32*512/4 float4s
        const int s = i >> 7, c4 = i & 127;
        *(float4*)&geneS[s * 516 + c4 * 4] = *(const float4*)&gb[s * 512 + c4 * 4];
    }
    const unsigned short* rb = rws + (size_t)b * 1024;
    for (int i = tid; i < 1024; i += 256) rS[i] = bf2f(rb[i]);

    if (tid < 32) {                                    // wave0: mask + invalid fix
        const bool pm = mask[b * 32 + tid] != 0;
        unsigned mb = (unsigned)__ballot(pm);
        const int inv = (mb == 0xffffffffu);
        if (inv) mb &= ~1u;                            // unmask slot 0
        if (tid == 0) { mbx[0] = mb; invalidws[b] = inv; }
    }
    __syncthreads();

    if (tid < 64) {                                    // waves 0,1: h = wave, s = lane
        const int h = tid >> 5, s = tid & 31;
        const float* gr = &geneS[s * 516];
        const float* rr = &rS[h * 512];
        float accv = 0.0f;
        for (int c = 0; c < 512; c += 4) {
            const float4 gg = *(const float4*)&gr[c];
            const float4 rv = *(const float4*)&rr[c];
            accv += gg.x * rv.x + gg.y * rv.y + gg.z * rv.z + gg.w * rv.w;
        }
        const unsigned mb = mbx[0];
        const bool pm = (mb >> s) & 1u;
        float sc = pm ? -INFINITY : accv;
        float mx = sc;
        for (int o = 16; o > 0; o >>= 1) mx = fmaxf(mx, __shfl_xor(mx, o, 32));
        const float e = pm ? 0.0f : __expf(sc - mx);
        float sum = e;
        for (int o = 16; o > 0; o >>= 1) sum += __shfl_xor(sum, o, 32);
        wS[h * 32 + s] = e / sum;
    }
    __syncthreads();

    if (tid < 32) attn_w[(size_t)b * 32 + tid] = 0.5f * (wS[tid] + wS[32 + tid]);

    {   // u[h, g] = sum_s w[h,s] * gene[s,g]  (4 cols/thread)
        const int h = tid >> 7, g4 = tid & 127;
        float4 u = {0, 0, 0, 0};
        for (int s = 0; s < 32; ++s) {
            const float w = wS[h * 32 + s];
            const float4 gg = *(const float4*)&geneS[s * 516 + g4 * 4];
            u.x += w * gg.x; u.y += w * gg.y; u.z += w * gg.z; u.w += w * gg.w;
        }
        ushort4 o;
        o.x = f2bf(u.x); o.y = f2bf(u.y); o.z = f2bf(u.z); o.w = f2bf(u.w);
        *(ushort4*)&uws[(size_t)b * 1024 + h * 512 + g4 * 4] = o;
    }
}

// ------------------------------- launcher ----------------------------------
extern "C" void kernel_launch(void* const* d_in, const int* in_sizes, int n_in,
                              void* d_out, int out_size, void* d_ws, size_t ws_size,
                              hipStream_t stream) {
    const float* source = (const float*)d_in[0];
    const float* gene   = (const float*)d_in[1];
    const int*   maskp  = (const int*)d_in[2];
    const float* mask_n = (const float*)d_in[3];
    const float* Wq     = (const float*)d_in[4];
    const float* bq     = (const float*)d_in[5];
    const float* Wk     = (const float*)d_in[6];
    // d_in[7] = bk: drops out of softmax (constant per (b,h))
    const float* Wv     = (const float*)d_in[8];
    const float* bv     = (const float*)d_in[9];
    const float* Wout   = (const float*)d_in[10];
    const float* bout   = (const float*)d_in[11];

    const int B = in_sizes[0] / 512;                  // 4096
    const int WN = 512 * 512;

    // workspace carve (bf16 stored as ushort)
    unsigned short* wq_b   = (unsigned short*)d_ws;
    unsigned short* wkT_b  = wq_b   + WN;
    unsigned short* wv_b   = wkT_b  + WN;
    unsigned short* wout_b = wv_b   + WN;
    unsigned short* src_b  = wout_b + WN;             // B*512
    unsigned short* q_b    = src_b  + (size_t)B * 512;
    unsigned short* r_b    = q_b    + (size_t)B * 512;   // B*1024
    unsigned short* u_b    = r_b    + (size_t)B * 1024;  // B*1024
    unsigned short* attn_b = u_b    + (size_t)B * 1024;  // B*512
    int*            inv_b  = (int*)(attn_b + (size_t)B * 512);

    float* outF  = (float*)d_out;
    float* attnW = outF + (size_t)B * 512;

    // prep: bf16 weights + source
    cvt_bf16_kernel<<<WN / 256, 256, 0, stream>>>(Wq, wq_b, WN);
    transpose_bf16_kernel<<<256, 256, 0, stream>>>(Wk, wkT_b);
    cvt_bf16_kernel<<<WN / 256, 256, 0, stream>>>(Wv, wv_b, WN);
    cvt_bf16_kernel<<<WN / 256, 256, 0, stream>>>(Wout, wout_b, WN);
    cvt_bf16_kernel<<<(B * 512) / 256, 256, 0, stream>>>(source, src_b, B * 512);

    const dim3 blk(256);
    // G1: q_s = (src @ Wq^T + bq) * hd^-0.5
    gemm_bf16_kernel<512, 0><<<dim3(B / 32, 2), blk, 0, stream>>>(
        src_b, 512, wq_b, 512, bq, 0.0625f, q_b, 512,
        nullptr, nullptr, nullptr, nullptr);
    // G2: r_h = q_s[:, h*256:...] @ Wk_h   (B-matrix via WkT, contiguous K)
    for (int h = 0; h < 2; ++h)
        gemm_bf16_kernel<256, 0><<<dim3(B / 32, 2), blk, 0, stream>>>(
            q_b + h * 256, 512, wkT_b + h * 256, 512, nullptr, 1.0f,
            r_b + h * 512, 1024, nullptr, nullptr, nullptr, nullptr);
    // A: scores + softmax + u + attn_weight  (single pass over gene)
    const size_t sh = (32 * 516 + 1024 + 64 + 4) * sizeof(float);
    attn_stream_kernel<<<B, blk, sh, stream>>>(gene, r_b, maskp, u_b, inv_b, attnW);
    // G3: attn_h = u_h @ Wv_h^T + bv_h
    for (int h = 0; h < 2; ++h)
        gemm_bf16_kernel<512, 0><<<dim3(B / 32, 1), blk, 0, stream>>>(
            u_b + h * 512, 1024, wv_b + (size_t)h * 256 * 512, 512,
            bv + h * 256, 1.0f, attn_b + h * 256, 512,
            nullptr, nullptr, nullptr, nullptr);
    // G4: out = gate(attn @ Wout^T + bout) * mask_n + src
    gemm_bf16_kernel<512, 1><<<dim3(B / 32, 2), blk, 0, stream>>>(
        attn_b, 512, wout_b, 512, bout, 1.0f, nullptr, 0,
        outF, source, mask_n, inv_b);
}
